// ReferenceLossAttnProcessor_90787018702866
// MI455X (gfx1250) — matrix operand order repaired and checked by reading, not compiled, not run on hardware
//
#include <hip/hip_runtime.h>

// ---------------------------------------------------------------------------
// MHA forward, B=2 S=4096 C=640 H=8 D=80 on gfx1250 (MI455X).
// ~113 GFLOP vs ~50MB HBM -> compute bound -> f16 WMMA (f32 accum) flash attn.
// CDNA5 data movement: double-buffered K/V tiles staged via
// GLOBAL_LOAD_ASYNC_TO_LDS_B128 (ASYNCcnt) and the Tensor Data Mover for K
// tiles (TENSORcnt), with __has_builtin guards + synchronous fallbacks.
// V is produced pre-transposed ([B,H,D,S]) by its projection GEMM so PV
// B-fragments are contiguous loads (no LDS scatter transpose).
// ---------------------------------------------------------------------------

constexpr int Bb = 2;
constexpr int Ss = 4096;
constexpr int Cc = 640;
constexpr int Hh = 8;
constexpr int Dd = 80;
constexpr int Dp = 96;            // head_dim padded to 3 x 32 WMMA K-chunks
constexpr int Mtot = Bb * Ss;     // 8192 tokens

typedef __attribute__((ext_vector_type(16))) _Float16 v16h;
typedef __attribute__((ext_vector_type(8)))  _Float16 h8;   // 16B
typedef __attribute__((ext_vector_type(4)))  _Float16 h4;   // 8B
typedef __attribute__((ext_vector_type(8)))  float    v8f;

#ifndef __has_builtin
#define __has_builtin(x) 0
#endif
#define HAS_ASYNC      __has_builtin(__builtin_amdgcn_global_load_async_to_lds_b128)
#define HAS_ASYNC_WAIT __has_builtin(__builtin_amdgcn_s_wait_asynccnt)
#define HAS_TDM        (__has_builtin(__builtin_amdgcn_tensor_load_to_lds) && \
                        __has_builtin(__builtin_amdgcn_s_wait_tensorcnt))

// Pointee type must match the builtin's declared parameter exactly:
// 'int __attribute__((vector_size(16)))' in AS(1)/AS(3).
typedef int v4i_vs __attribute__((vector_size(16)));
typedef __attribute__((address_space(1))) v4i_vs gas_v4i;
typedef __attribute__((address_space(3))) v4i_vs las_v4i;
typedef __attribute__((address_space(3))) void   las_void;

// ---- async copy primitive: 16B global -> LDS (per-lane addresses) ----------
__device__ __forceinline__ void cp16_async(const _Float16* g, _Float16* l) {
#if HAS_ASYNC
    __builtin_amdgcn_global_load_async_to_lds_b128((gas_v4i*)g, (las_v4i*)l, 0, 0);
#else
    *(h8*)l = *(const h8*)g;                       // synchronous fallback
#endif
}

#if !HAS_ASYNC
#define ASYNC_WAIT(n)
#elif HAS_ASYNC_WAIT
#define ASYNC_WAIT(n) __builtin_amdgcn_s_wait_asynccnt(n)
#else
#define ASYNC_WAIT(n) asm volatile("s_wait_asynccnt " #n ::: "memory")
#endif

#if HAS_TDM
#define TENSOR_WAIT(n) __builtin_amdgcn_s_wait_tensorcnt(n)
// async ops per tile = 3 (V only); K goes through the TDM
#define WAIT_PROLOGUE() ASYNC_WAIT(3)
#define WAIT_TILE()     do { ASYNC_WAIT(3); TENSOR_WAIT(1); } while (0)
#define WAIT_LAST()     do { ASYNC_WAIT(0); TENSOR_WAIT(0); } while (0)
#else
#define TENSOR_WAIT(n)
// async ops per tile = 6 (K:3 + V:3)
#define WAIT_PROLOGUE() ASYNC_WAIT(6)
#define WAIT_TILE()     ASYNC_WAIT(6)
#define WAIT_LAST()     ASYNC_WAIT(0)
#endif

#if HAS_TDM
typedef __attribute__((ext_vector_type(4))) unsigned int u32x4;
typedef __attribute__((ext_vector_type(8))) int          i32x8;
typedef __attribute__((ext_vector_type(4))) int          i32x4;
// One TDM descriptor moves a 64x96 f16 K tile. tensor_dim0=96 < tile_dim0=104
// exploits documented OOB-reads-return-zero to fill the +8-half LDS row pad,
// yielding conflict-free 104-half LDS row stride for the b128 fragment loads.
__device__ __forceinline__ void tdm_load_k(const _Float16* gsrc, _Float16* ldst) {
    unsigned lds = (unsigned)(unsigned long long)(las_void*)ldst;   // LDS byte offset
    unsigned long long ga = (unsigned long long)gsrc;               // global byte addr
    u32x4 g0 = { 1u /*count*/, lds, (unsigned)ga,
                 (unsigned)(ga >> 32) | (2u << 30) /*type=image*/ };
    i32x8 g1 = { (int)(1u << 16),     // data_size = 2 bytes
                 (int)(96u << 16),    // tensor_dim0 = 96 (lo16 @ [31:16])
                 (int)(4096u << 16),  // tensor_dim0 hi = 0 | tensor_dim1 = 4096
                 (int)(104u << 16),   // tensor_dim1 hi = 0 | tile_dim0 = 104
                 64,                  // tile_dim1 = 64 | tile_dim2 = 0
                 96,                  // tensor_dim0_stride = 96 (lo32)
                 0, 0 };
    i32x4 z4 = { 0, 0, 0, 0 };
    i32x8 z8 = { 0, 0, 0, 0, 0, 0, 0, 0 };
    __builtin_amdgcn_tensor_load_to_lds(g0, g1, z4, z4, z8, 0);
}
#endif

// D = A(16x32 f16) * B(32x16 f16) + C(16x16 f32)
__device__ __forceinline__ v8f wmma16(v16h a, v16h b, v8f c) {
    return __builtin_amdgcn_wmma_f32_16x16x32_f16(false, a, false, b,
                                                  (short)0, c, false, false);
}

// A-matrix fragment (16x32 f16) from row-major LDS [row][k].
__device__ __forceinline__ v16h load_a(const _Float16* p0, int stride, int k0) {
    const int lane = threadIdx.x & 31;
    const int r = lane & 15, hf = lane >> 4;
    const _Float16* p = p0 + r * stride + k0 + 8 * hf;
    h8 lo = *(const h8*)p;
    h8 hi = *(const h8*)(p + 16);
    v16h f;
#pragma unroll
    for (int i = 0; i < 8; ++i) { f[i] = lo[i]; f[i + 8] = hi[i]; }
    return f;
}

// B-matrix fragment (32x16 f16) from B^T row-major LDS [n][k].
__device__ __forceinline__ v16h load_b(const _Float16* p0, int stride, int k0) {
    const int lane = threadIdx.x & 31;
    const int n = lane & 15, hf = lane >> 4;
    const _Float16* p = p0 + n * stride + k0 + 8 * hf;
    h8 lo = *(const h8*)p;
    h8 hi = *(const h8*)(p + 16);
    v16h f;
#pragma unroll
    for (int i = 0; i < 8; ++i) { f[i] = lo[i]; f[i + 8] = hi[i]; }
    return f;
}

// Zero the d=80..95 pad columns of the Q/K f16 workspace (idempotent).
__global__ void pad_init(_Float16* q, _Float16* k) {
    int t = blockIdx.x * blockDim.x + threadIdx.x;     // 0 .. B*H*S-1
    size_t o = (size_t)t * Dp + Dd;
    h8 z = {};
    *(h8*)(q + o) = z; *(h8*)(q + o + 8) = z;
    *(h8*)(k + o) = z; *(h8*)(k + o + 8) = z;
}

// ---------------------------------------------------------------------------
// Projection GEMM: f16 out = X[8192,640] @ W[640,640] + b
// VT=false: out[b,h,s,d] stride Dp (Q,K).  VT=true: out[b,h,d,s] (V^T).
// ---------------------------------------------------------------------------
template <bool VT>
__global__ __launch_bounds__(256) void qkv_gemm(const float* __restrict__ X,
                                                const float* __restrict__ W,
                                                const float* __restrict__ bias,
                                                _Float16* __restrict__ out) {
    __shared__ _Float16 sX[128][40];   // 32 + 8 pad: 16B aligned, conflict-free
    __shared__ _Float16 sW[64][40];    // W^T tile [n][k]
    const int tid = threadIdx.x, lane = tid & 31, wave = tid >> 5;
    const int hf = lane >> 4, cl = lane & 15;
    const int m0 = blockIdx.x * 128, n0 = blockIdx.y * 64;

    v8f acc[4] = {};
    for (int kt = 0; kt < Cc; kt += 32) {
#pragma unroll
        for (int it = 0; it < 4; ++it) {            // X tile 128x32 f32 -> f16
            int f = tid + it * 256;
            int r = f >> 3, c4 = (f & 7) << 2;
            float4 xv = *(const float4*)(X + (size_t)(m0 + r) * Cc + kt + c4);
            h4 hv = {(_Float16)xv.x, (_Float16)xv.y, (_Float16)xv.z, (_Float16)xv.w};
            *(h4*)&sX[r][c4] = hv;
        }
#pragma unroll
        for (int it = 0; it < 2; ++it) {            // W tile 32x64, transposed
            int f = tid + it * 256;
            int kr = f >> 4, nc = (f & 15) << 2;
            float4 wv = *(const float4*)(W + (size_t)(kt + kr) * Cc + n0 + nc);
            sW[nc + 0][kr] = (_Float16)wv.x;
            sW[nc + 1][kr] = (_Float16)wv.y;
            sW[nc + 2][kr] = (_Float16)wv.z;
            sW[nc + 3][kr] = (_Float16)wv.w;
        }
        if (kt + 32 < Cc)                           // global_prefetch_b8
            __builtin_prefetch(X + (size_t)(m0 + (tid >> 3)) * Cc + kt + 32, 0, 1);
        __syncthreads();
        v16h a = load_a(&sX[wave * 16][0], 40, 0);
#pragma unroll
        for (int nb = 0; nb < 4; ++nb)
            acc[nb] = wmma16(a, load_b(&sW[nb * 16][0], 40, 0), acc[nb]);
        __syncthreads();
    }
#pragma unroll
    for (int nb = 0; nb < 4; ++nb) {
        int col = n0 + nb * 16 + cl;
        int head = col / Dd, d = col % Dd;
        float bv = bias[col];
#pragma unroll
        for (int i = 0; i < 8; ++i) {
            int row = m0 + wave * 16 + i + 8 * hf;
            int bb = row >> 12, ss = row & (Ss - 1);
            float v = acc[nb][i] + bv;
            if (VT)   // V^T: [b,h,d,s]
                out[((size_t)((bb * Hh + head) * Dd + d)) * Ss + ss] = (_Float16)v;
            else      // Q/K: [b,h,s,d] stride Dp
                out[((size_t)((bb * Hh + head) * Ss + ss)) * Dp + d] = (_Float16)v;
        }
    }
}

// ---------------------------------------------------------------------------
// Flash attention: one (b,h) x 128-query tile per WG; 64-key blocks,
// double-buffered LDS tiles streamed by async-to-LDS (+ TDM for K).
// ---------------------------------------------------------------------------
__global__ __launch_bounds__(256) void attn_fwd(const _Float16* __restrict__ Q,
                                                const _Float16* __restrict__ K,
                                                const _Float16* __restrict__ Vt,
                                                _Float16* __restrict__ O) {
    __shared__ _Float16 sQ[128][104];       // 26.6 KB
    __shared__ _Float16 sK[2][64][104];     // 26.6 KB (double buffered)
    __shared__ _Float16 sVt[2][80][72];     // 23.0 KB  [d][key], 64+8 pad
    __shared__ _Float16 sP[8][16][72];      // 18.4 KB  per-wave probs [m][key]
    const int tid = threadIdx.x, lane = tid & 31, wave = tid >> 5;
    const int hf = lane >> 4, cl = lane & 15;
    const int q0 = blockIdx.x * 128;
    const int bh = blockIdx.y;
    const _Float16* qg = Q + (size_t)bh * Ss * Dp;
    const _Float16* kg = K + (size_t)bh * Ss * Dp;
    const _Float16* vg = Vt + (size_t)bh * Dd * Ss;
    const float scale = 0.11180339887498948f;      // 1/sqrt(80)

    // ---- issue Q tile (6 async ops/wave) then tile 0 --------------------
#pragma unroll
    for (int it = 0; it < 6; ++it) {
        int f = tid + it * 256;
        int r = f / 12, c8 = (f % 12) * 8;
        cp16_async(qg + (size_t)(q0 + r) * Dp + c8, &sQ[r][c8]);
    }
    {   // tile 0 -> buffer 0
        const _Float16* kt0 = kg;            // key block 0
#if HAS_TDM
        if (wave == 0) tdm_load_k(kt0, &sK[0][0][0]);
#else
#pragma unroll
        for (int it = 0; it < 3; ++it) {
            int f = tid + it * 256;
            int r = f / 12, c8 = (f % 12) * 8;
            cp16_async(kt0 + (size_t)r * Dp + c8, &sK[0][r][c8]);
        }
#endif
#pragma unroll
        for (int it = 0; it < 3; ++it) {
            int f = tid + it * 256;
            int r = f >> 3; r = r > 79 ? 79 : r;     // duplicate, same data
            int c8 = (f & 7) << 3;
            cp16_async(vg + (size_t)r * Ss + c8, &sVt[0][r][c8]);
        }
    }
    WAIT_PROLOGUE();                 // own-wave Q writes landed (in-order)
    __syncthreads();                 // everyone's Q writes landed
    v16h qf[3];
#pragma unroll
    for (int c = 0; c < 3; ++c) qf[c] = load_a(&sQ[wave * 16][0], 104, c * 32);

    v8f oacc[5] = {};
    float Mr[8], Lr[8];
#pragma unroll
    for (int i = 0; i < 8; ++i) { Mr[i] = -1e30f; Lr[i] = 0.0f; }

    for (int t = 0; t < Ss / 64; ++t) {
        const int cur = t & 1;
        if (t < Ss / 64 - 1) {       // stream tile t+1 into the other buffer
            const int nxt = cur ^ 1;
            const _Float16* kn = kg + (size_t)(t + 1) * 64 * Dp;
            const _Float16* vn = vg + (size_t)(t + 1) * 64;
#if HAS_TDM
            if (wave == 0) tdm_load_k(kn, &sK[nxt][0][0]);
#else
#pragma unroll
            for (int it = 0; it < 3; ++it) {
                int f = tid + it * 256;
                int r = f / 12, c8 = (f % 12) * 8;
                cp16_async(kn + (size_t)r * Dp + c8, &sK[nxt][r][c8]);
            }
#endif
#pragma unroll
            for (int it = 0; it < 3; ++it) {
                int f = tid + it * 256;
                int r = f >> 3; r = r > 79 ? 79 : r;
                int c8 = (f & 7) << 3;
                cp16_async(vn + (size_t)r * Ss + c8, &sVt[nxt][r][c8]);
            }
            WAIT_TILE();             // current tile landed (next still inflight)
        } else {
            WAIT_LAST();
        }
        __syncthreads();

        // Scores: 16x64 per wave, K=96 -> 12 WMMA
        v8f sacc[4] = {};
#pragma unroll
        for (int nb = 0; nb < 4; ++nb) {
            v16h b0 = load_b(&sK[cur][nb * 16][0], 104, 0);
            v16h b1 = load_b(&sK[cur][nb * 16][0], 104, 32);
            v16h b2 = load_b(&sK[cur][nb * 16][0], 104, 64);
            sacc[nb] = wmma16(qf[0], b0, sacc[nb]);
            sacc[nb] = wmma16(qf[1], b1, sacc[nb]);
            sacc[nb] = wmma16(qf[2], b2, sacc[nb]);
        }

        // Online softmax (C layout: slot i, lane -> row i+8*hf, col nb*16+cl)
#pragma unroll
        for (int i = 0; i < 8; ++i) {
            float mx = fmaxf(fmaxf(sacc[0][i], sacc[1][i]),
                             fmaxf(sacc[2][i], sacc[3][i]));
#pragma unroll
            for (int m = 8; m >= 1; m >>= 1) mx = fmaxf(mx, __shfl_xor(mx, m, 32));
            float mnew = fmaxf(Mr[i], mx * scale);
            float rs = 0.0f;
#pragma unroll
            for (int nb = 0; nb < 4; ++nb) {
                float e = __expf(sacc[nb][i] * scale - mnew);
                sP[wave][i + 8 * hf][nb * 16 + cl] = (_Float16)e;
                rs += e;
            }
#pragma unroll
            for (int m = 8; m >= 1; m >>= 1) rs += __shfl_xor(rs, m, 32);
            float alpha = __expf(Mr[i] - mnew);
            Lr[i] = Lr[i] * alpha + rs;
            Mr[i] = mnew;
#pragma unroll
            for (int db = 0; db < 5; ++db) oacc[db][i] *= alpha;
        }

        // O += P @ V : 2 K-chunks x 5 D-blocks = 10 WMMA
#pragma unroll
        for (int c = 0; c < 2; ++c) {
            v16h aP = load_a(&sP[wave][0][0], 72, c * 32);
#pragma unroll
            for (int db = 0; db < 5; ++db)
                oacc[db] = wmma16(aP, load_b(&sVt[cur][db * 16][0], 72, c * 32),
                                  oacc[db]);
        }
        __syncthreads();             // all waves done with buffer `cur`
    }

    // Normalize, write attn output [B,S,C] f16 (feeds output GEMM)
    const int b = bh >> 3, h = bh & 7;
#pragma unroll
    for (int i = 0; i < 8; ++i) {
        int srow = q0 + wave * 16 + i + 8 * hf;
        float inv = 1.0f / Lr[i];
#pragma unroll
        for (int db = 0; db < 5; ++db) {
            int d = db * 16 + cl;
            O[((size_t)(b * Ss + srow)) * Cc + h * Dd + d] =
                (_Float16)(oacc[db][i] * inv);
        }
    }
}

// ---------------------------------------------------------------------------
// Output projection: out(fp32) = attn(f16) @ Wo + bo + residual
// ---------------------------------------------------------------------------
__global__ __launch_bounds__(256) void out_gemm(const _Float16* __restrict__ A,
                                                const float* __restrict__ W,
                                                const float* __restrict__ bias,
                                                const float* __restrict__ resid,
                                                float* __restrict__ out) {
    __shared__ _Float16 sX[128][40];
    __shared__ _Float16 sW[64][40];
    const int tid = threadIdx.x, lane = tid & 31, wave = tid >> 5;
    const int hf = lane >> 4, cl = lane & 15;
    const int m0 = blockIdx.x * 128, n0 = blockIdx.y * 64;

    v8f acc[4] = {};
    for (int kt = 0; kt < Cc; kt += 32) {
#pragma unroll
        for (int it = 0; it < 2; ++it) {            // A tile (already f16)
            int f = tid + it * 256;
            int r = f >> 2, c8 = (f & 3) << 3;
            *(h8*)&sX[r][c8] = *(const h8*)(A + (size_t)(m0 + r) * Cc + kt + c8);
        }
#pragma unroll
        for (int it = 0; it < 2; ++it) {            // W tile, transposed
            int f = tid + it * 256;
            int kr = f >> 4, nc = (f & 15) << 2;
            float4 wv = *(const float4*)(W + (size_t)(kt + kr) * Cc + n0 + nc);
            sW[nc + 0][kr] = (_Float16)wv.x;
            sW[nc + 1][kr] = (_Float16)wv.y;
            sW[nc + 2][kr] = (_Float16)wv.z;
            sW[nc + 3][kr] = (_Float16)wv.w;
        }
        if (kt + 32 < Cc)
            __builtin_prefetch(A + (size_t)(m0 + (tid >> 1)) * Cc + kt + 32, 0, 1);
        __syncthreads();
        v16h a = load_a(&sX[wave * 16][0], 40, 0);
#pragma unroll
        for (int nb = 0; nb < 4; ++nb)
            acc[nb] = wmma16(a, load_b(&sW[nb * 16][0], 40, 0), acc[nb]);
        __syncthreads();
    }
#pragma unroll
    for (int nb = 0; nb < 4; ++nb) {
        int col = n0 + nb * 16 + cl;
        float bv = bias[col];
#pragma unroll
        for (int i = 0; i < 8; ++i) {
            int row = m0 + wave * 16 + i + 8 * hf;
            size_t o = (size_t)row * Cc + col;
            out[o] = acc[nb][i] + bv + resid[o];
        }
    }
}

// ---------------------------------------------------------------------------
extern "C" void kernel_launch(void* const* d_in, const int* in_sizes, int n_in,
                              void* d_out, int out_size, void* d_ws, size_t ws_size,
                              hipStream_t stream) {
    const float* X  = (const float*)d_in[0];
    const float* Wq = (const float*)d_in[1];
    const float* bq = (const float*)d_in[2];
    const float* Wk = (const float*)d_in[3];
    const float* bk = (const float*)d_in[4];
    const float* Wv = (const float*)d_in[5];
    const float* bv = (const float*)d_in[6];
    const float* Wo = (const float*)d_in[7];
    const float* bo = (const float*)d_in[8];

    _Float16* ws = (_Float16*)d_ws;
    const size_t QKSZ = (size_t)Bb * Hh * Ss * Dp;   // Q,K buffers (padded D)
    const size_t VSZ  = (size_t)Bb * Hh * Dd * Ss;   // V^T buffer
    _Float16* qh   = ws;
    _Float16* kh   = ws + QKSZ;
    _Float16* vh   = ws + 2 * QKSZ;
    _Float16* attn = ws + 2 * QKSZ + VSZ;            // [B,S,C] f16

    pad_init<<<(Bb * Hh * Ss) / 256, 256, 0, stream>>>(qh, kh);

    dim3 g(Mtot / 128, Cc / 64);
    qkv_gemm<false><<<g, 256, 0, stream>>>(X, Wq, bq, qh);
    qkv_gemm<false><<<g, 256, 0, stream>>>(X, Wk, bk, kh);
    qkv_gemm<true ><<<g, 256, 0, stream>>>(X, Wv, bv, vh);

    attn_fwd<<<dim3(Ss / 128, Bb * Hh), 256, 0, stream>>>(qh, kh, vh, attn);

    out_gemm<<<g, 256, 0, stream>>>(attn, Wo, bo, X, (float*)d_out);
}
